// ImprovedSpeMamba_43954695307651
// MI455X (gfx1250) — matrix operand order repaired and checked by reading, not compiled, not run on hardware
//
#include <hip/hip_runtime.h>
#include <hip/hip_bf16.h>
#include <math.h>

// ---------------- constants (match reference) ----------------
#define CCH   128      // channels
#define TOK   4        // token_num
#define GRP   32       // d_model
#define DIN   64       // d_inner
#define DST   16       // d_state
#define HW    16384    // H*W
#define PIXB  2097152  // per-image floats = CCH*HW
#define NPIX  32768    // B*H*W
#define BLKPIX 16      // pixels per block (4 per wave, 4 waves)

typedef __attribute__((ext_vector_type(16))) _Float16 v16h;
typedef __attribute__((ext_vector_type(8)))  float    v8f;

// ---------------- WMMA helpers (CDNA5 wave32 layouts) ----------------
// A (16x32 f16): lane<16 holds M=lane, K=(i&7)+16*(i>>3); lane>=16 adds +8 to K.
__device__ __forceinline__ v16h a_frag_lds(const float* A, int ld, int kofs, int lane) {
  const int m  = lane & 15;
  const int kb = ((lane >> 4) << 3) + kofs;
  v16h a;
#pragma unroll
  for (int i = 0; i < 16; ++i) {
    const int k = (i & 7) + kb + ((i >> 3) << 4);
    a[i] = (_Float16)A[m * ld + k];
  }
  return a;
}

// B (32x16 f16) from global weight W[nout][kdim] (row-major, out-major):
// lane holds N=(lane&15)+16*ntile, K = (i&7) + 16*(i>>3) + 8*(lane>>4) + kofs.
__device__ __forceinline__ v16h b_frag_gw(const float* __restrict__ W, int nout,
                                          int kdim, int ntile, int kofs, int lane) {
  const int n  = (lane & 15) + (ntile << 4);
  const int kb = ((lane >> 4) << 3) + kofs;
  v16h b;
#pragma unroll
  for (int i = 0; i < 16; ++i) {
    const float v = (n < nout) ? W[n * kdim + kb + (i & 7) + ((i >> 3) << 4)] : 0.0f;
    b[i] = (_Float16)v;
  }
  return b;
}

__device__ __forceinline__ v8f wmma_f16(v16h a, v16h b, v8f c) {
  return __builtin_amdgcn_wmma_f32_16x16x32_f16(false, a, false, b, (short)0, c,
                                                false, false);
}

__device__ __forceinline__ float silu_f(float v) { return v / (1.f + __expf(-v)); }

// ---------------- K1: fused per-pixel Mamba ----------------
__global__ __launch_bounds__(128) void mamba_k1(
    const float* __restrict__ x,
    const float* __restrict__ in_proj_w,   // (128,32)
    const float* __restrict__ conv_w,      // (64,4)
    const float* __restrict__ conv_b,      // (64)
    const float* __restrict__ x_proj_w,    // (34,64)
    const float* __restrict__ dt_proj_w,   // (64,2)
    const float* __restrict__ dt_proj_b,   // (64)
    const float* __restrict__ A_log,       // (64,16)
    const float* __restrict__ Dvec,        // (64)
    const float* __restrict__ out_proj_w,  // (32,64)
    float* __restrict__ xr,                // (2,128,128,128) -> lives in d_out
    float* __restrict__ S1, float* __restrict__ S2) // (2,128) each
{
  // sShared overlaid: [stage0] per-wave 16x32 A-matrix (4*512) ; [stage6] 128x16 xr tile
  __shared__ __align__(16) float    sShared[2048];
  __shared__ __align__(16) float    sXc [4][16 * 64];  // conv output (f32)
  __shared__ __align__(16) float    sDbl[4][16 * 36];  // dt(2)|B(16)|C(16)
  __shared__ __align__(16) float    sTmp[4][16 * 64];  // xi -> delta -> gated y
  __shared__ __align__(16) _Float16 sZ  [4][16 * 64];  // silu(z)

  const int tid  = threadIdx.x;
  const int w    = tid >> 5;
  const int lane = tid & 31;
  const int pixbase = blockIdx.x * BLKPIX;       // 16 consecutive pixels, same image
  const int b       = pixbase >> 14;
  const int hwbase  = pixbase & (HW - 1);
  const size_t xbase = (size_t)b * PIXB + hwbase;
  const int pixw = w * 4;                        // this wave's first local pixel

  // ---- stage 0: stage this wave's A-matrix (16 rows x 32) ----
  float* Xin = &sShared[w * 512];
  for (int idx = lane; idx < 512; idx += 32) {
    const int m = idx >> 5, k = idx & 31;
    const int c = ((m & 3) << 5) | k;            // token*32 + k
    Xin[idx] = x[xbase + (size_t)c * HW + (pixw + (m >> 2))];
  }
  __syncthreads();

  // ---- stage 1: in_proj (8 WMMAs), split xi / silu(z) ----
  {
    const v16h a = a_frag_lds(Xin, 32, 0, lane);
#pragma unroll
    for (int j = 0; j < 8; ++j) {
      const v16h bw = b_frag_gw(in_proj_w, 128, 32, j, 0, lane);
      v8f acc = {};
      acc = wmma_f16(a, bw, acc);
#pragma unroll
      for (int r = 0; r < 8; ++r) {
        const int m  = r + ((lane >> 4) << 3);
        const int nl = lane & 15;
        if (j < 4) sTmp[w][m * 64 + (j << 4) + nl] = acc[r];                  // xi
        else       sZ[w][m * 64 + ((j - 4) << 4) + nl] = (_Float16)silu_f(acc[r]);
      }
    }
  }
  __syncthreads();

  // ---- stage 2: causal depthwise conv1d over tokens + SiLU ----
  for (int idx = lane; idx < 1024; idx += 32) {
    const int m = idx >> 6, d = idx & 63;
    const int p = m >> 2, t = m & 3;
    float acc = conv_b[d];
#pragma unroll
    for (int k = 0; k < 4; ++k) {
      const int tt = t + k - 3;
      if (tt >= 0) acc += conv_w[d * 4 + k] * sTmp[w][(4 * p + tt) * 64 + d];
    }
    sXc[w][m * 64 + d] = silu_f(acc);
  }
  __syncthreads();

  // ---- stage 3: x_proj (K=64 -> 2 chained WMMAs x 3 N-tiles, 34 valid) ----
  {
    const v16h a0 = a_frag_lds(sXc[w], 64, 0,  lane);
    const v16h a1 = a_frag_lds(sXc[w], 64, 32, lane);
#pragma unroll
    for (int jt = 0; jt < 3; ++jt) {
      v8f acc = {};
      acc = wmma_f16(a0, b_frag_gw(x_proj_w, 34, 64, jt, 0,  lane), acc);
      acc = wmma_f16(a1, b_frag_gw(x_proj_w, 34, 64, jt, 32, lane), acc);
#pragma unroll
      for (int r = 0; r < 8; ++r) {
        const int m   = r + ((lane >> 4) << 3);
        const int col = (jt << 4) + (lane & 15);
        if (col < 34) sDbl[w][m * 36 + col] = acc[r];
      }
    }
  }
  __syncthreads();

  // ---- stage 4: dt_proj + softplus (overwrite xi region with delta) ----
  for (int idx = lane; idx < 1024; idx += 32) {
    const int m = idx >> 6, d = idx & 63;
    const float v = sDbl[w][m * 36 + 0] * dt_proj_w[d * 2 + 0]
                  + sDbl[w][m * 36 + 1] * dt_proj_w[d * 2 + 1] + dt_proj_b[d];
    sTmp[w][m * 64 + d] = (v > 20.f) ? v : log1pf(__expf(v));
  }
  __syncthreads();

  // ---- stage 5: selective scan (L=4), gate by silu(z), overwrite delta with y ----
#pragma unroll
  for (int k = 0; k < 8; ++k) {
    const int pair = (k << 5) | lane;   // 256 (pixel,d) pairs per wave
    const int p = pair >> 6, d = pair & 63;
    float Arow[16];
#pragma unroll
    for (int s = 0; s < 16; ++s) Arow[s] = -__expf(A_log[d * 16 + s]);
    const float Dd = Dvec[d];
    float h[16];
#pragma unroll
    for (int s = 0; s < 16; ++s) h[s] = 0.f;
#pragma unroll
    for (int t = 0; t < 4; ++t) {
      const int r = 4 * p + t;
      const float dlt = sTmp[w][r * 64 + d];
      const float u   = sXc[w][r * 64 + d];
      const float du  = dlt * u;
      float acc = 0.f;
#pragma unroll
      for (int s = 0; s < 16; ++s) {
        const float dA = __expf(dlt * Arow[s]);
        h[s] = dA * h[s] + du * sDbl[w][r * 36 + 2 + s];
        acc += h[s] * sDbl[w][r * 36 + 18 + s];
      }
      const float y = acc + u * Dd;
      sTmp[w][r * 64 + d] = y * (float)sZ[w][r * 64 + d];   // gated
    }
  }
  __syncthreads();

  // ---- stage 6: out_proj (K=64 -> 2x2 WMMAs), stage xr tile [c][pix] ----
  {
    const v16h a0 = a_frag_lds(sTmp[w], 64, 0,  lane);
    const v16h a1 = a_frag_lds(sTmp[w], 64, 32, lane);
#pragma unroll
    for (int jn = 0; jn < 2; ++jn) {
      v8f acc = {};
      acc = wmma_f16(a0, b_frag_gw(out_proj_w, 32, 64, jn, 0,  lane), acc);
      acc = wmma_f16(a1, b_frag_gw(out_proj_w, 32, 64, jn, 32, lane), acc);
#pragma unroll
      for (int r = 0; r < 8; ++r) {
        const int m = r + ((lane >> 4) << 3);
        const int t = m & 3, p = m >> 2;
        const int c = t * 32 + (jn << 4) + (lane & 15);
        sShared[c * 16 + (pixw + p)] = acc[r];
      }
    }
  }
  __syncthreads();

  // ---- stage 7: coalesced store + per-channel sum/sumsq for SE/GN ----
  {
    const int c = tid;                           // 128 threads == 128 channels
    float s1 = 0.f, s2 = 0.f;
    float* dst = xr + xbase + (size_t)c * HW;
#pragma unroll
    for (int pl = 0; pl < 16; pl += 4) {
      const float4 v = *(const float4*)&sShared[c * 16 + pl];
      s1 += v.x + v.y + v.z + v.w;
      s2 += v.x * v.x + v.y * v.y + v.z * v.z + v.w * v.w;
      *(float4*)(dst + pl) = v;
    }
    atomicAdd(&S1[b * CCH + c], s1);
    atomicAdd(&S2[b * CCH + c], s2);
  }
}

// ---------------- K2: SE MLP + analytic GroupNorm -> per-(b,c) affine ----------------
__global__ __launch_bounds__(128) void se_gn_k2(
    const float* __restrict__ S1, const float* __restrict__ S2,
    const float* __restrict__ se1_w,  // (32,128)
    const float* __restrict__ se2_w,  // (128,32)
    const float* __restrict__ gn_w, const float* __restrict__ gn_b,
    float* __restrict__ alpha, float* __restrict__ beta)
{
  __shared__ float pooled[2][128], t1[2][32], gsh[2][128], mus[2][4], isd[2][4];
  const int tid = threadIdx.x;
  for (int bb = 0; bb < 2; ++bb) pooled[bb][tid] = S1[bb * 128 + tid] * (1.f / HW);
  __syncthreads();
  if (tid < 32) {
    for (int bb = 0; bb < 2; ++bb) {
      float a = 0.f;
      for (int c = 0; c < 128; ++c) a += pooled[bb][c] * se1_w[tid * 128 + c];
      t1[bb][tid] = a / (1.f + __expf(-a));                 // SiLU
    }
  }
  __syncthreads();
  for (int bb = 0; bb < 2; ++bb) {
    float a = 0.f;
    for (int j = 0; j < 32; ++j) a += t1[bb][j] * se2_w[tid * 32 + j];
    gsh[bb][tid] = 1.f / (1.f + __expf(-a));                // sigmoid -> g
  }
  __syncthreads();
  if (tid < 8) {
    const int bb = tid >> 2, g = tid & 3;
    float m = 0.f, e2 = 0.f;
    for (int c = g * 32; c < g * 32 + 32; ++c) {
      const float gv = gsh[bb][c];
      m  += gv * S1[bb * 128 + c];
      e2 += gv * gv * S2[bb * 128 + c];
    }
    m  *= 1.f / (32.f * HW);
    e2 *= 1.f / (32.f * HW);
    mus[bb][g] = m;
    isd[bb][g] = rsqrtf(fmaxf(e2 - m * m, 0.f) + 1e-5f);
  }
  __syncthreads();
  for (int bb = 0; bb < 2; ++bb) {
    const int g = tid >> 5;
    const float is = isd[bb][g], wv = gn_w[tid];
    alpha[bb * 128 + tid] = gsh[bb][tid] * is * wv;
    beta [bb * 128 + tid] = gn_b[tid] - mus[bb][g] * is * wv;
  }
}

// ---------------- K3: out = silu(xr*alpha + beta) + x (in-place on d_out) ----------------
__global__ __launch_bounds__(256) void apply_k3(
    float* __restrict__ out, const float* __restrict__ x,
    const float* __restrict__ alpha, const float* __restrict__ beta)
{
  const size_t i4 = ((size_t)blockIdx.x * blockDim.x + threadIdx.x) * 4;
  if (i4 >= (size_t)2 * PIXB) return;
  const int bb = (int)(i4 >> 21);
  const int c  = (int)((i4 >> 14) & 127);
  const float al = alpha[bb * 128 + c], be = beta[bb * 128 + c];
  float4 v  = *(const float4*)(out + i4);
  const float4 xi = *(const float4*)(x + i4);
  float t;
  t = v.x * al + be; v.x = t / (1.f + __expf(-t)) + xi.x;
  t = v.y * al + be; v.y = t / (1.f + __expf(-t)) + xi.y;
  t = v.z * al + be; v.z = t / (1.f + __expf(-t)) + xi.z;
  t = v.w * al + be; v.w = t / (1.f + __expf(-t)) + xi.w;
  *(float4*)(out + i4) = v;
}

// ---------------- launcher ----------------
extern "C" void kernel_launch(void* const* d_in, const int* in_sizes, int n_in,
                              void* d_out, int out_size, void* d_ws, size_t ws_size,
                              hipStream_t stream) {
  const float* x          = (const float*)d_in[0];
  const float* in_proj_w  = (const float*)d_in[1];
  const float* conv_w     = (const float*)d_in[2];
  const float* conv_b     = (const float*)d_in[3];
  const float* x_proj_w   = (const float*)d_in[4];
  const float* dt_proj_w  = (const float*)d_in[5];
  const float* dt_proj_b  = (const float*)d_in[6];
  const float* A_log      = (const float*)d_in[7];
  const float* Dvec       = (const float*)d_in[8];
  const float* out_proj_w = (const float*)d_in[9];
  const float* se1_w      = (const float*)d_in[10];
  const float* se2_w      = (const float*)d_in[11];
  const float* gn_w       = (const float*)d_in[12];
  const float* gn_b       = (const float*)d_in[13];

  float* xr = (float*)d_out;          // mamba output staged in d_out, finished in-place by K3
  float* ws = (float*)d_ws;
  float* S1    = ws;                  // (2,128)
  float* S2    = ws + 256;            // (2,128)
  float* alpha = ws + 512;            // (2,128)
  float* beta  = ws + 768;            // (2,128)

  hipMemsetAsync(S1, 0, 512 * sizeof(float), stream);   // capture-safe

  mamba_k1<<<NPIX / BLKPIX, 128, 0, stream>>>(
      x, in_proj_w, conv_w, conv_b, x_proj_w, dt_proj_w, dt_proj_b,
      A_log, Dvec, out_proj_w, xr, S1, S2);

  se_gn_k2<<<1, 128, 0, stream>>>(S1, S2, se1_w, se2_w, gn_w, gn_b, alpha, beta);

  apply_k3<<<(2 * PIXB / 4 + 255) / 256, 256, 0, stream>>>(xr, x, alpha, beta);
}
